// GemNetT_15281493639525
// MI455X (gfx1250) — compile-verified, matrix operand order checked by reading, not run on hardware
//
#include <hip/hip_runtime.h>
#include <hip/hip_bf16.h>
#include <math.h>

// ---------------------------------------------------------------------------
// GemNet-T forward for MI455X (gfx1250, wave32, WMMA).
// Dense layers -> one bf16-WMMA tiled GEMM (v_wmma_f32_16x16x32_bf16, f32
// accumulate), weights pre-converted to bf16 [N][K] on device. Grid places
// N-tiles on blockIdx.x so column-blocks sharing an A stripe are co-resident
// and A re-reads hit the 192MB L2 (workload is HBM-bandwidth bound).
// Irregular graph ops (triplet scatter, segment sums) are f32 atomics.
// ---------------------------------------------------------------------------

#define INV_SQRT2F 0.70710678118654752f

typedef __attribute__((ext_vector_type(16))) __bf16 v16bf;
typedef __attribute__((ext_vector_type(8)))  __bf16 v8bf;
typedef __attribute__((ext_vector_type(8)))  float  v8f;

__device__ __forceinline__ __bf16 to_bf16(float f) {
  union { float f; unsigned u; } v; v.f = f;
  unsigned r = v.u + 0x7fffu + ((v.u >> 16) & 1u);   // round-to-nearest-even
  union { unsigned short s; __bf16 b; } o; o.s = (unsigned short)(r >> 16);
  return o.b;
}

__device__ __forceinline__ float ssilu(float x) {
  // gemnet "swish" = SiLU / 0.6
  return x * (1.0f / (1.0f + __expf(-x))) * (1.0f / 0.6f);
}

// ---------------------------------------------------------------------------
// WMMA GEMM: C[M,N] = epilogue(A[M,K] @ Wt^T)   with Wt = bf16 [N][K]
//   epilogue: v = acc; if(act) v=ssilu(v); if(resid) v=(resid+v)*scale;
//             if(emul) v*=emul;
//   A f32 (ABF16=false) or bf16 (ABF16=true), row-major, lda == K.
// Tile BM=128 x BN=64 x BK=32; 8 wave32 waves (4Mx2N), each wave 32x32 via
// 2x2 v_wmma_f32_16x16x32_bf16.
// ---------------------------------------------------------------------------
#define BM 128
#define BN 64
#define BK 32
#define LDSK 40   // padded K stride (80B rows: 16B-aligned, spreads banks)

template <bool ABF16>
__global__ __launch_bounds__(256)
void k_gemm(const void* __restrict__ Ap, const __bf16* __restrict__ Wt,
            float* __restrict__ C, int M, int N, int K,
            int act, const float* __restrict__ resid, float scale,
            const float* __restrict__ emul)
{
  __shared__ __align__(16) __bf16 As[BM][LDSK];
  __shared__ __align__(16) __bf16 Bs[BN][LDSK];

  const int tid     = threadIdx.x;
  const int colBase = blockIdx.x * BN;   // N-tiles fast-varying (L2 reuse of A)
  const int rowBase = blockIdx.y * BM;
  const int wid  = tid >> 5;
  const int lane = tid & 31;
  const int wm = wid & 3;        // wave row (0..3) -> 32 rows each
  const int wn = wid >> 2;       // wave col (0..1) -> 32 cols each
  const int g  = lane >> 4;      // lane group (K-half select)
  const int q  = lane & 15;      // row (A) / col (B,C) within 16

  const bool fullM = (rowBase + BM) <= M;

  v8f acc[2][2];
#pragma unroll
  for (int a = 0; a < 2; ++a)
#pragma unroll
    for (int b = 0; b < 2; ++b)
#pragma unroll
      for (int r = 0; r < 8; ++r) acc[a][b][r] = 0.0f;

  for (int k0 = 0; k0 < K; k0 += BK) {
    const bool fullK = (k0 + BK) <= K;
    // ---- stage A: chunks of 8 k-elements (16B LDS stores) ----
    if (fullM && fullK) {
      if (ABF16) {
        const __bf16* A = (const __bf16*)Ap;
#pragma unroll
        for (int c = tid; c < (BM * BK) / 8; c += 256) {
          int r = c >> 2, kc = (c & 3) * 8;
          const __bf16* src = A + (size_t)(rowBase + r) * K + (k0 + kc);
          *(v8bf*)&As[r][kc] = *(const v8bf*)src;
          if (k0 + BK < K) __builtin_prefetch(src + BK, 0, 1);
        }
      } else {
        const float* A = (const float*)Ap;
#pragma unroll
        for (int c = tid; c < (BM * BK) / 8; c += 256) {
          int r = c >> 2, kc = (c & 3) * 8;
          const float* src = A + (size_t)(rowBase + r) * K + (k0 + kc);
          float4 a0 = *(const float4*)src;
          float4 a1 = *(const float4*)(src + 4);
          v8bf v;
          v[0] = to_bf16(a0.x); v[1] = to_bf16(a0.y);
          v[2] = to_bf16(a0.z); v[3] = to_bf16(a0.w);
          v[4] = to_bf16(a1.x); v[5] = to_bf16(a1.y);
          v[6] = to_bf16(a1.z); v[7] = to_bf16(a1.w);
          *(v8bf*)&As[r][kc] = v;
          if (k0 + BK < K) __builtin_prefetch(src + BK, 0, 1);
        }
      }
    } else {  // edge tile: masked scalar path
      for (int i = tid; i < BM * BK; i += 256) {
        int r = i >> 5, k = i & 31;
        int gr = rowBase + r, gk = k0 + k;
        float v = 0.0f;
        if (gr < M && gk < K)
          v = ABF16 ? (float)((const __bf16*)Ap)[(size_t)gr * K + gk]
                    : ((const float*)Ap)[(size_t)gr * K + gk];
        As[r][k] = to_bf16(v);
      }
    }
    // ---- stage B from bf16 [N][K]: contiguous 16B chunks ----
#pragma unroll
    for (int c = tid; c < (BN * BK) / 8; c += 256) {
      int n = c >> 2, kc = (c & 3) * 8;
      int gn = colBase + n, gk = k0 + kc;
      if (gn < N && (gk + 8) <= K) {
        *(v8bf*)&Bs[n][kc] = *(const v8bf*)(Wt + (size_t)gn * K + gk);
      } else {
        v8bf z;
#pragma unroll
        for (int j = 0; j < 8; ++j)
          z[j] = (gn < N && (gk + j) < K) ? Wt[(size_t)gn * K + gk + j] : to_bf16(0.0f);
        *(v8bf*)&Bs[n][kc] = z;
      }
    }
    __syncthreads();

    // ---- fragments: 16-bit WMMA layout, lane group g holds K ranges
    //      [8g,8g+8) (elems 0..7) and [16+8g,16+8g+8) (elems 8..15) ----
    v16bf afr[2], bfr[2];
#pragma unroll
    for (int mt = 0; mt < 2; ++mt) {
      const __bf16* ar = &As[wm * 32 + mt * 16 + q][0];
      v8bf lo = *(const v8bf*)(ar + 8 * g);
      v8bf hi = *(const v8bf*)(ar + 16 + 8 * g);
#pragma unroll
      for (int i = 0; i < 8; ++i) { afr[mt][i] = lo[i]; afr[mt][8 + i] = hi[i]; }
    }
#pragma unroll
    for (int nt = 0; nt < 2; ++nt) {
      const __bf16* br = &Bs[wn * 32 + nt * 16 + q][0];
      v8bf lo = *(const v8bf*)(br + 8 * g);
      v8bf hi = *(const v8bf*)(br + 16 + 8 * g);
#pragma unroll
      for (int i = 0; i < 8; ++i) { bfr[nt][i] = lo[i]; bfr[nt][8 + i] = hi[i]; }
    }
#pragma unroll
    for (int mt = 0; mt < 2; ++mt)
#pragma unroll
      for (int nt = 0; nt < 2; ++nt)
        acc[mt][nt] = __builtin_amdgcn_wmma_f32_16x16x32_bf16(
            false, afr[mt], false, bfr[nt], (short)0, acc[mt][nt], false, false);
    __syncthreads();
  }

  // ---- epilogue: C element r of lane (g,q) is (M = r + 8g, N = q) ----
#pragma unroll
  for (int mt = 0; mt < 2; ++mt)
#pragma unroll
    for (int nt = 0; nt < 2; ++nt) {
      int gc = colBase + wn * 32 + nt * 16 + q;
#pragma unroll
      for (int r = 0; r < 8; ++r) {
        int gr = rowBase + wm * 32 + mt * 16 + 8 * g + r;
        if (gr < M && gc < N) {
          size_t o = (size_t)gr * N + gc;
          float v = acc[mt][nt][r];
          if (act) v = ssilu(v);
          if (resid) v = (resid[o] + v) * scale;
          if (emul) v *= emul[o];
          C[o] = v;
        }
      }
    }
}

// ---------------------------------------------------------------------------
// Weight prep: Wt[n*K+k] = bf16(W[k*N+n])   (reads coalesced)
// ---------------------------------------------------------------------------
__global__ void k_prep_w(const float* __restrict__ W, __bf16* __restrict__ Wt,
                         int K, int N) {
  size_t i = (size_t)blockIdx.x * blockDim.x + threadIdx.x;
  if (i >= (size_t)K * N) return;
  int k = (int)(i / N), n = (int)(i % N);
  Wt[(size_t)n * K + k] = to_bf16(W[i]);
}

// cbf: Wt[(s*16+c)*128 + r] = bf16(mlp_cbf3[s,r,c])  -> [N=112][K=128]
__global__ void k_prep_cbf(const float* __restrict__ src, __bf16* __restrict__ dst) {
  int i = blockIdx.x * blockDim.x + threadIdx.x;
  if (i >= 7 * 128 * 16) return;
  int s = i / (128 * 16), r = (i / 16) % 128, c = i % 16;
  dst[(s * 16 + c) * 128 + r] = to_bf16(src[i]);
}

// wbil: Wt[o*1024 + (c*64+t)] = bf16(Wbil[t,c,o])  -> [N=64][K=1024]
__global__ void k_prep_wbil(const float* __restrict__ src, __bf16* __restrict__ dst) {
  int i = blockIdx.x * blockDim.x + threadIdx.x;
  if (i >= 64 * 16 * 64) return;
  int t = i / (16 * 64), c = (i / 64) % 16, o = i % 64;
  dst[o * 1024 + c * 64 + t] = to_bf16(src[i]);
}

// ---------------------------------------------------------------------------
// Small kernels
// ---------------------------------------------------------------------------
__global__ void k_zero4(float4* p, size_t n4) {
  size_t i = (size_t)blockIdx.x * blockDim.x + threadIdx.x;
  if (i < n4) p[i] = make_float4(0.f, 0.f, 0.f, 0.f);
}

// rbf[e, r] : gaussian RBF * polynomial envelope (p=5)
__global__ void k_radial(const float* __restrict__ D, float* __restrict__ rbf, size_t n) {
  size_t i = (size_t)blockIdx.x * blockDim.x + threadIdx.x;
  if (i >= n) return;
  size_t e = i >> 7; int r = (int)(i & 127);
  float d = D[e] * (1.0f / 6.0f);
  float d2 = d * d, d5 = d2 * d2 * d;
  float env = (d < 1.0f) ? (1.0f + d5 * (-21.0f + d * (35.0f + d * (-15.0f)))) : 0.0f;
  float off = (float)r * (1.0f / 127.0f);
  float coeff = -0.5f * 127.0f * 127.0f;
  float x = d - off;
  rbf[i] = env * __expf(coeff * x * x);
}

// sph[t, l] : m=0 real spherical harmonics via Legendre recurrence
__global__ void k_sph(const float* __restrict__ V, const int* __restrict__ ca,
                      const int* __restrict__ ba, float* __restrict__ sph, int T3) {
  int t = blockIdx.x * blockDim.x + threadIdx.x;
  if (t >= T3) return;
  int ea = ca[t], eb = ba[t];
  float c = V[ea * 3 + 0] * V[eb * 3 + 0] + V[ea * 3 + 1] * V[eb * 3 + 1] +
            V[ea * 3 + 2] * V[eb * 3 + 2];
  const float inv4pi = 0.07957747154594767f;  // 1/(4*pi)
  float Pm2 = 1.0f, Pm1 = c;
  sph[(size_t)t * 7 + 0] = Pm2 * sqrtf(1.0f * inv4pi);
  sph[(size_t)t * 7 + 1] = Pm1 * sqrtf(3.0f * inv4pi);
#pragma unroll
  for (int l = 2; l < 7; ++l) {
    float P = ((2.0f * l - 1.0f) * c * Pm1 - (l - 1.0f) * Pm2) / (float)l;
    sph[(size_t)t * 7 + l] = P * sqrtf((2.0f * l + 1.0f) * inv4pi);
    Pm2 = Pm1; Pm1 = P;
  }
}

__global__ void k_gather_h(const float* __restrict__ emb, const int* __restrict__ an,
                           float* __restrict__ h, size_t n) {
  size_t i = (size_t)blockIdx.x * blockDim.x + threadIdx.x;
  if (i >= n) return;
  size_t a = i >> 7; int j = (int)(i & 127);
  h[i] = emb[(size_t)an[a] * 128 + j];
}

// out[e, 0:128]=h[idx_s], [128:256]=h[idx_t], [256:]=tail[e]; width = 256+tailW
__global__ void k_cat3(const float* __restrict__ h, const float* __restrict__ tail,
                       const int* __restrict__ is, const int* __restrict__ it,
                       float* __restrict__ out, int tailW, size_t n) {
  size_t i = (size_t)blockIdx.x * blockDim.x + threadIdx.x;
  if (i >= n) return;
  int W = 256 + tailW;
  size_t e = i / W; int j = (int)(i - e * W);
  float v;
  if (j < 128)       v = h[(size_t)is[e] * 128 + j];
  else if (j < 256)  v = h[(size_t)it[e] * 128 + (j - 128)];
  else               v = tail[e * (size_t)tailW + (j - 256)];
  out[i] = v;
}

__global__ void k_addscale(float* __restrict__ out, const float* __restrict__ a,
                           const float* __restrict__ b, float s, size_t n) {
  size_t i = (size_t)blockIdx.x * blockDim.x + threadIdx.x;
  if (i < n) out[i] = (a[i] + b[i]) * s;
}

// out[e][j] = (a[e][j] + b[e^1][j]) * s  (width 256)
__global__ void k_swapadd(float* __restrict__ out, const float* __restrict__ a,
                          const float* __restrict__ b, float s, size_t n) {
  size_t i = (size_t)blockIdx.x * blockDim.x + threadIdx.x;
  if (i >= n) return;
  size_t e = i >> 8; int j = (int)(i & 255);
  out[i] = (a[i] + b[((e ^ 1) << 8) + j]) * s;
}

// acc[idx_t[e]][j] += m[e][j] * w[e][j]  (width 256)
__global__ void k_scatter_mw(const float* __restrict__ m, const float* __restrict__ w,
                             const int* __restrict__ it, float* __restrict__ acc, size_t n) {
  size_t i = (size_t)blockIdx.x * blockDim.x + threadIdx.x;
  if (i >= n) return;
  size_t e = i >> 8; int j = (int)(i & 255);
  atomicAdd(&acc[(size_t)it[e] * 256 + j], m[i] * w[i]);
}

// acc[row] += X[row,:] . w  : one wave32 per row, shfl butterfly reduce
__global__ __launch_bounds__(256)
void k_rowdot_accum(const float* __restrict__ X, const float* __restrict__ w,
                    float* __restrict__ acc, int M, int K) {
  __shared__ float ws[256];
  for (int i = threadIdx.x; i < K; i += blockDim.x) ws[i] = w[i];
  __syncthreads();
  int lane = threadIdx.x & 31;
  int row = blockIdx.x * 8 + (threadIdx.x >> 5);
  if (row >= M) return;
  const float* x = X + (size_t)row * K;
  float s = 0.0f;
  for (int k = lane; k < K; k += 32) s += x[k] * ws[k];
#pragma unroll
  for (int off = 16; off > 0; off >>= 1) s += __shfl_xor(s, off, 32);
  if (lane == 0) acc[row] += s;
}

// sumk[id3_ca[t]][s*64+ch] += sph[t][s] * x64[id3_ba[t]][ch]   n = T3*64
__global__ void k_scatter_trip(const float* __restrict__ sph, const float* __restrict__ x64,
                               const int* __restrict__ ba, const int* __restrict__ ca,
                               float* __restrict__ sumk, size_t n) {
  size_t i = (size_t)blockIdx.x * blockDim.x + threadIdx.x;
  if (i >= n) return;
  size_t t = i >> 6; int ch = (int)(i & 63);
  float xv = x64[(size_t)ba[t] * 64 + ch];
  float* dst = sumk + (size_t)ca[t] * 448 + ch;
  const float* sp = sph + t * 7;
#pragma unroll
  for (int s = 0; s < 7; ++s) atomicAdd(dst + s * 64, sp[s] * xv);
}

// X[e, c*64+t] = sum_s rbfW1[e, s*16+c] * sumk[e, s*64+t]   (bf16 out) n = E*1024
__global__ void k_tmpX(const float* __restrict__ rbfW1, const float* __restrict__ sumk,
                       __bf16* __restrict__ X, size_t n) {
  size_t i = (size_t)blockIdx.x * blockDim.x + threadIdx.x;
  if (i >= n) return;
  size_t e = i >> 10; int ct = (int)(i & 1023);
  int c = ct >> 6, t = ct & 63;
  const float* rw = rbfW1 + e * 112;
  const float* sk = sumk + e * 448;
  float acc = 0.0f;
#pragma unroll
  for (int s = 0; s < 7; ++s) acc += rw[s * 16 + c] * sk[s * 64 + t];
  X[i] = to_bf16(acc);
}

__global__ void k_segE(const float* __restrict__ Et, const int* __restrict__ batch,
                       float* __restrict__ out, int NA) {
  int a = blockIdx.x * blockDim.x + threadIdx.x;
  if (a >= NA) return;
  atomicAdd(&out[batch[a]], Et[a]);
}

__global__ void k_scatF(const float* __restrict__ Fst, const float* __restrict__ V,
                        const int* __restrict__ it, float* __restrict__ outF, int NE) {
  int e = blockIdx.x * blockDim.x + threadIdx.x;
  if (e >= NE) return;
  float f = Fst[e];
  int a = it[e];
#pragma unroll
  for (int k = 0; k < 3; ++k) atomicAdd(&outF[(size_t)a * 3 + k], f * V[(size_t)e * 3 + k]);
}

// ---------------------------------------------------------------------------
// Host-side orchestration
// ---------------------------------------------------------------------------
struct OutBlk {
  const float *dense_rbf, *dense1, *res[6], *out_energy;
  const float *f_dense1, *f_res[6], *dense_rbf_F, *out_forces;
};
struct IntBlk {
  const float *dense_ca, *dense_ba, *trip_mlp_rbf, *down_proj, *Wbil;
  const float *up_ca, *up_ac, *before[2], *after[4];
  const float *au_dense_rbf, *au_dense1, *au_res[6], *concat, *res_m[2];
};

static inline unsigned gblk(size_t n) { return (unsigned)((n + 255) / 256); }

struct Ctx {
  hipStream_t s;
  __bf16* wscratch;   // bf16 [N][K] weight staging (max 512*256)
};

// prep weight (f32 KxN -> bf16 NxK) then GEMM
static void gemm(Ctx& c, const void* A, bool abf16, const float* W, float* Cout,
                 int M, int N, int K, int act, const float* resid, float scale,
                 const float* emul) {
  k_prep_w<<<gblk((size_t)K * N), 256, 0, c.s>>>(W, c.wscratch, K, N);
  dim3 grid((unsigned)((N + BN - 1) / BN), (unsigned)((M + BM - 1) / BM));
  if (abf16) k_gemm<true ><<<grid, 256, 0, c.s>>>(A, c.wscratch, Cout, M, N, K, act, resid, scale, emul);
  else       k_gemm<false><<<grid, 256, 0, c.s>>>(A, c.wscratch, Cout, M, N, K, act, resid, scale, emul);
}

// GEMM with already-prepped bf16 [N][K] weight
static void gemmT(Ctx& c, const void* A, bool abf16, const __bf16* Wt, float* Cout,
                  int M, int N, int K, int act, const float* resid, float scale,
                  const float* emul) {
  dim3 grid((unsigned)((N + BN - 1) / BN), (unsigned)((M + BM - 1) / BM));
  if (abf16) k_gemm<true ><<<grid, 256, 0, c.s>>>(A, Wt, Cout, M, N, K, act, resid, scale, emul);
  else       k_gemm<false><<<grid, 256, 0, c.s>>>(A, Wt, Cout, M, N, K, act, resid, scale, emul);
}

// X = (X + act(act(X@W1)@W2)) * 1/sqrt(2), optional final *= emul
static void residual2(Ctx& c, float* X, const float* W1, const float* W2,
                      float* T, int M, int D, const float* emul = nullptr) {
  gemm(c, X, false, W1, T, M, D, D, 1, nullptr, 0.0f, nullptr);
  gemm(c, T, false, W2, X, M, D, D, 1, X, INV_SQRT2F, emul);
}

extern "C" void kernel_launch(void* const* d_in, const int* in_sizes, int n_in,
                              void* d_out, int out_size, void* d_ws, size_t ws_size,
                              hipStream_t stream) {
  // ---- inputs (setup_inputs() dict insertion order) ----
  const int*   atomic_numbers = (const int*)  d_in[0];
  const int*   batch          = (const int*)  d_in[1];
  const int*   idx_s          = (const int*)  d_in[2];
  const int*   idx_t          = (const int*)  d_in[3];
  const float* D_st           = (const float*)d_in[4];
  const float* V_st           = (const float*)d_in[5];
  const int*   id3_ba         = (const int*)  d_in[6];
  const int*   id3_ca         = (const int*)  d_in[7];

  const int NA = in_sizes[0];       // 10000 atoms
  const int NE = in_sizes[2];       // 250000 edges
  const int T3 = in_sizes[6];       // 800000 triplets

  // ---- params: leaves flattened recursively in dict-insertion / list order ----
  int pi = 8;
  auto P = [&]() -> const float* { return (const float*)d_in[pi++]; };
  const float* atom_emb    = P();
  const float* edge_emb    = P();
  const float* mlp_rbf3    = P();
  const float* mlp_cbf3    = P();
  const float* mlp_rbf_h   = P();
  const float* mlp_rbf_out = P();
  IntBlk ib[3];
  for (int i = 0; i < 3; ++i) {
    ib[i].dense_ca = P(); ib[i].dense_ba = P(); ib[i].trip_mlp_rbf = P();
    ib[i].down_proj = P(); ib[i].Wbil = P(); ib[i].up_ca = P(); ib[i].up_ac = P();
    for (int j = 0; j < 2; ++j) ib[i].before[j] = P();
    for (int j = 0; j < 4; ++j) ib[i].after[j]  = P();
    ib[i].au_dense_rbf = P(); ib[i].au_dense1 = P();
    for (int j = 0; j < 6; ++j) ib[i].au_res[j] = P();
    ib[i].concat = P();
    for (int j = 0; j < 2; ++j) ib[i].res_m[j] = P();
  }
  OutBlk ob[4];
  for (int i = 0; i < 4; ++i) {
    ob[i].dense_rbf = P(); ob[i].dense1 = P();
    for (int j = 0; j < 6; ++j) ob[i].res[j] = P();
    ob[i].out_energy = P(); ob[i].f_dense1 = P();
    for (int j = 0; j < 6; ++j) ob[i].f_res[j] = P();
    ob[i].dense_rbf_F = P(); ob[i].out_forces = P();
  }
  (void)n_in;

  // ---- workspace carving ----
  char* wp = (char*)d_ws;
  auto carve = [&](size_t bytes) -> char* {
    char* p = wp; wp += (bytes + 255) & ~(size_t)255; return p;
  };
  float*  rbf     = (float*) carve((size_t)NE * 128 * 4);
  float*  rbf3    = (float*) carve((size_t)NE * 16 * 4);
  float*  rbf_h   = (float*) carve((size_t)NE * 16 * 4);
  float*  rbf_out = (float*) carve((size_t)NE * 16 * 4);
  float*  rbfW1   = (float*) carve((size_t)NE * 112 * 4);
  float*  sph     = (float*) carve((size_t)T3 * 7 * 4);
  float*  h       = (float*) carve((size_t)NA * 128 * 4);
  float*  m       = (float*) carve((size_t)NE * 256 * 4);
  float*  cat     = (float*) carve((size_t)NE * 512 * 4);
  float*  bufA    = (float*) carve((size_t)NE * 256 * 4);
  float*  bufB    = (float*) carve((size_t)NE * 256 * 4);
  float*  bufC    = (float*) carve((size_t)NE * 256 * 4);
  float*  bufD    = (float*) carve((size_t)NE * 256 * 4);
  float*  x64a    = (float*) carve((size_t)NE * 64 * 4);
  float*  x64b    = (float*) carve((size_t)NE * 64 * 4);
  float*  sumk    = (float*) carve((size_t)NE * 448 * 4);
  __bf16* Xbf     = (__bf16*)carve((size_t)NE * 1024 * 2);
  float*  atomacc = (float*) carve((size_t)NA * 256 * 4);
  float*  xa      = (float*) carve((size_t)NA * 128 * 4);
  float*  xt      = (float*) carve((size_t)NA * 128 * 4);
  float*  E_t     = (float*) carve((size_t)NA * 4);
  float*  F_st    = (float*) carve((size_t)NE * 4);
  __bf16* cbfT    = (__bf16*)carve((size_t)112 * 128 * 2);
  __bf16* wbilT   = (__bf16*)carve((size_t)64 * 1024 * 2);
  __bf16* wscr    = (__bf16*)carve((size_t)512 * 256 * 2);
  (void)ws_size;

  float* outE = (float*)d_out;                 // (NUM_MOLS) energies
  float* outF = outE + (out_size - 3 * NA);    // (NA, 3) forces

  Ctx c{stream, wscr};
  auto zero = [&](float* p, size_t n) {  // n multiple of 4
    k_zero4<<<gblk(n / 4), 256, 0, stream>>>((float4*)p, n / 4);
  };

  // ---- basis / embedding precompute ----
  zero((float*)d_out, (size_t)out_size);
  zero(E_t, (size_t)NA);
  zero(F_st, (size_t)NE);
  k_radial<<<gblk((size_t)NE * 128), 256, 0, stream>>>(D_st, rbf, (size_t)NE * 128);
  k_sph<<<gblk((size_t)T3), 256, 0, stream>>>(V_st, id3_ca, id3_ba, sph, T3);
  k_prep_cbf<<<gblk(7 * 128 * 16), 256, 0, stream>>>(mlp_cbf3, cbfT);
  gemm(c, rbf, false, mlp_rbf3,    rbf3,    NE, 16, 128, 0, nullptr, 0.f, nullptr);
  gemm(c, rbf, false, mlp_rbf_h,   rbf_h,   NE, 16, 128, 0, nullptr, 0.f, nullptr);
  gemm(c, rbf, false, mlp_rbf_out, rbf_out, NE, 16, 128, 0, nullptr, 0.f, nullptr);
  gemmT(c, rbf, false, cbfT, rbfW1, NE, 112, 128, 0, nullptr, 0.f, nullptr);
  k_gather_h<<<gblk((size_t)NA * 128), 256, 0, stream>>>(atom_emb, atomic_numbers, h,
                                                         (size_t)NA * 128);
  k_cat3<<<gblk((size_t)NE * 384), 256, 0, stream>>>(h, rbf, idx_s, idx_t, cat, 128,
                                                     (size_t)NE * 384);
  gemm(c, cat, false, edge_emb, m, NE, 256, 384, 1, nullptr, 0.f, nullptr);  // m

  // ---- output block (energy head into E_t, force head into F_st) ----
  auto run_out_block = [&](const OutBlk& o) {
    // energy side
    gemm(c, rbf_out, false, o.dense_rbf, bufC, NE, 256, 16, 0, nullptr, 0.f, nullptr);
    zero(atomacc, (size_t)NA * 256);
    k_scatter_mw<<<gblk((size_t)NE * 256), 256, 0, stream>>>(m, bufC, idx_t, atomacc,
                                                             (size_t)NE * 256);
    gemm(c, atomacc, false, o.dense1, xa, NA, 128, 256, 1, nullptr, 0.f, nullptr);
    for (int r = 0; r < 3; ++r)
      residual2(c, xa, o.res[2 * r], o.res[2 * r + 1], xt, NA, 128);
    k_rowdot_accum<<<gblk((size_t)NA * 32), 256, 0, stream>>>(xa, o.out_energy, E_t, NA, 128);
    // force side: last residual's epilogue fuses the rbf gate multiply
    gemm(c, rbf_out, false, o.dense_rbf_F, bufC, NE, 256, 16, 0, nullptr, 0.f, nullptr);
    gemm(c, m, false, o.f_dense1, bufA, NE, 256, 256, 1, nullptr, 0.f, nullptr);
    residual2(c, bufA, o.f_res[0], o.f_res[1], bufB, NE, 256);
    residual2(c, bufA, o.f_res[2], o.f_res[3], bufB, NE, 256);
    residual2(c, bufA, o.f_res[4], o.f_res[5], bufB, NE, 256, bufC);
    k_rowdot_accum<<<gblk((size_t)NE * 32), 256, 0, stream>>>(bufA, o.out_forces, F_st, NE, 256);
  };

  run_out_block(ob[0]);

  // ---- interaction blocks ----
  for (int blk = 0; blk < 3; ++blk) {
    const IntBlk& b = ib[blk];
    // triplet branch: x_ba = act(m@dense_ba) * (rbf3@trip_mlp_rbf)  (fused emul)
    gemm(c, rbf3, false, b.trip_mlp_rbf, bufC, NE, 256, 16, 0, nullptr, 0.f, nullptr);
    gemm(c, m, false, b.dense_ba, bufA, NE, 256, 256, 1, nullptr, 0.f, bufC);
    gemm(c, bufA, false, b.down_proj, x64a, NE, 64, 256, 1, nullptr, 0.f, nullptr);
    // segment-sum over triplets: sumk[E, 7, 64]
    zero(sumk, (size_t)NE * 448);
    k_scatter_trip<<<gblk((size_t)T3 * 64), 256, 0, stream>>>(sph, x64a, id3_ba, id3_ca,
                                                              sumk, (size_t)T3 * 64);
    // tmp[e,c,t] (bf16) then bilinear GEMM: xbil = X(E,1024) @ Wbil'(1024,64)
    k_tmpX<<<gblk((size_t)NE * 1024), 256, 0, stream>>>(rbfW1, sumk, Xbf, (size_t)NE * 1024);
    k_prep_wbil<<<gblk(64 * 16 * 64), 256, 0, stream>>>(b.Wbil, wbilT);
    gemmT(c, Xbf, true, wbilT, x64b, NE, 64, 1024, 0, nullptr, 0.f, nullptr);
    // x3 = (act(xbil@up_ca) + act(xbil@up_ac)[id_swap]) / sqrt(2)
    gemm(c, x64b, false, b.up_ca, bufA, NE, 256, 64, 1, nullptr, 0.f, nullptr);
    gemm(c, x64b, false, b.up_ac, bufC, NE, 256, 64, 1, nullptr, 0.f, nullptr);
    k_swapadd<<<gblk((size_t)NE * 256), 256, 0, stream>>>(bufD, bufA, bufC, INV_SQRT2F,
                                                          (size_t)NE * 256);
    // x = (act(m@dense_ca) + x3)/sqrt(2); before-skip residual
    gemm(c, m, false, b.dense_ca, bufA, NE, 256, 256, 1, bufD, INV_SQRT2F, nullptr);
    residual2(c, bufA, b.before[0], b.before[1], bufB, NE, 256);
    // m = (m + x)/sqrt(2); after-skip residuals
    k_addscale<<<gblk((size_t)NE * 256), 256, 0, stream>>>(m, m, bufA, INV_SQRT2F,
                                                           (size_t)NE * 256);
    residual2(c, m, b.after[0], b.after[1], bufB, NE, 256);
    residual2(c, m, b.after[2], b.after[3], bufB, NE, 256);
    // atom update
    gemm(c, rbf_h, false, b.au_dense_rbf, bufC, NE, 256, 16, 0, nullptr, 0.f, nullptr);
    zero(atomacc, (size_t)NA * 256);
    k_scatter_mw<<<gblk((size_t)NE * 256), 256, 0, stream>>>(m, bufC, idx_t, atomacc,
                                                             (size_t)NE * 256);
    gemm(c, atomacc, false, b.au_dense1, xa, NA, 128, 256, 1, nullptr, 0.f, nullptr);
    for (int r = 0; r < 3; ++r)
      residual2(c, xa, b.au_res[2 * r], b.au_res[2 * r + 1], xt, NA, 128);
    k_addscale<<<gblk((size_t)NA * 128), 256, 0, stream>>>(h, h, xa, INV_SQRT2F,
                                                           (size_t)NA * 128);
    // concat edge update: m2 = act([h_s, h_t, m] @ concat); res; m = (m+m2)/sqrt(2)
    k_cat3<<<gblk((size_t)NE * 512), 256, 0, stream>>>(h, m, idx_s, idx_t, cat, 256,
                                                       (size_t)NE * 512);
    gemm(c, cat, false, b.concat, bufA, NE, 256, 512, 1, nullptr, 0.f, nullptr);
    residual2(c, bufA, b.res_m[0], b.res_m[1], bufB, NE, 256);
    k_addscale<<<gblk((size_t)NE * 256), 256, 0, stream>>>(m, m, bufA, INV_SQRT2F,
                                                           (size_t)NE * 256);

    run_out_block(ob[blk + 1]);
  }

  // ---- final reductions into d_out ----
  k_segE<<<gblk((size_t)NA), 256, 0, stream>>>(E_t, batch, outE, NA);
  k_scatF<<<gblk((size_t)NE), 256, 0, stream>>>(F_st, V_st, idx_t, outF, NE);
}
